// ESN_292057776184
// MI455X (gfx1250) — matrix operand order repaired and checked by reading, not compiled, not run on hardware
//
#include <hip/hip_runtime.h>
#include <hip/hip_bf16.h>
#include <math.h>

#define HIDDEN   2048
#define T_STEPS  8192
#define WASHOUT  200
#define NBLK     64
#define ROWS_PER_WG (HIDDEN / NBLK)   /* 32 */
#define THREADS  256

typedef __attribute__((ext_vector_type(16))) __bf16 v16bf;
typedef __attribute__((ext_vector_type(8)))  __bf16 v8bf;
typedef __attribute__((ext_vector_type(8)))  float  v8f;
typedef __attribute__((ext_vector_type(4)))  float  v4f;

// ---------------------------------------------------------------------------
// Workspace layout (bytes):
//   [0, 256)                         : grid-barrier counter (uint)
//   [256, 256 + 2*HIDDEN*4)          : x state double buffer (f32)
//   [256 + 16384, + TOUT*NBLK*4)     : per-(t, wg) partial readout dots
// ---------------------------------------------------------------------------

__global__ void esn_init(unsigned int* cnt, float* x0) {
  if (threadIdx.x == 0) *cnt = 0u;
  for (int i = threadIdx.x; i < HIDDEN; i += THREADS) x0[i] = 0.0f;
}

__global__ __launch_bounds__(THREADS)
void esn_recur(const float* __restrict__ u,
               const float* __restrict__ w_in,
               const float* __restrict__ w_res,
               const float* __restrict__ w_out,
               float* __restrict__ xstate,    // [2][HIDDEN]
               float* __restrict__ partials,  // [T-WASHOUT][NBLK]
               unsigned int* __restrict__ cnt)
{
  __shared__ __attribute__((aligned(64))) __bf16 xbf[HIDDEN];
  __shared__ float red[8][16];          // [tile*4 + kslice][row-in-tile]
  __shared__ float dotbuf[ROWS_PER_WG];

  const int tid   = threadIdx.x;
  const int wg    = blockIdx.x;
  const int lane  = tid & 31;
  const int wave  = tid >> 5;
  const int tile  = wave >> 2;          // 0..1  (16-row M tile)
  const int ks    = wave & 3;           // 0..3  (512-wide K slice)
  const int hi    = lane >> 4;          // half-wave group
  const int m     = lane & 15;
  const int khalf = hi * 8;
  const int row0  = wg * ROWS_PER_WG;
  const int arow  = row0 + tile * 16 + m;

  // ---- Build bf16 A fragments in registers, once (16 x 512 slice / wave) ----
  // A 16x32 layout: lane holds row m; VGPR0-3 = K khalf..khalf+7,
  // VGPR4-7 = K 16+khalf..16+khalf+7 (khalf = 0 lo-lanes / 8 hi-lanes).
  v16bf a[16];
#pragma unroll
  for (int c = 0; c < 16; ++c) {
    const int kc = ks * 512 + c * 32;
    const float* p = w_res + (size_t)arow * HIDDEN + kc + khalf;
    v4f f0 = *(const v4f*)(p);
    v4f f1 = *(const v4f*)(p + 4);
    v4f f2 = *(const v4f*)(p + 16);
    v4f f3 = *(const v4f*)(p + 20);
#pragma unroll
    for (int j = 0; j < 4; ++j) {
      a[c][j]      = (__bf16)f0[j];
      a[c][4 + j]  = (__bf16)f1[j];
      a[c][8 + j]  = (__bf16)f2[j];
      a[c][12 + j] = (__bf16)f3[j];
    }
  }

  // Per-row constants for the activation stage (threads 0..31 own one row each)
  float wIn = 0.0f, wOut = 0.0f;
  if (tid < ROWS_PER_WG) {
    wIn  = w_in[row0 + tid];
    wOut = w_out[row0 + tid];
  }

  for (int t = 0; t < T_STEPS; ++t) {
    const float* xprev = xstate + (size_t)(t & 1) * HIDDEN;
    float*       xnext = xstate + (size_t)((t + 1) & 1) * HIDDEN;

    // Broadcast x_{t-1} into LDS as bf16: each thread owns 8 contiguous
    // elements -> two global b128 loads + one ds_store_b128, no exec loop.
    {
      const v4f* xp4 = (const v4f*)xprev;
      v4f g0 = xp4[tid * 2 + 0];
      v4f g1 = xp4[tid * 2 + 1];
      v8bf pk;
#pragma unroll
      for (int j = 0; j < 4; ++j) {
        pk[j]     = (__bf16)g0[j];
        pk[4 + j] = (__bf16)g1[j];
      }
      *(v8bf*)(xbf + tid * 8) = pk;
    }
    __syncthreads();

    // y_rows += W_slice * x_slice via WMMA (x replicated across B columns).
    // B 32x16 layout: lane holds K=kc+khalf..+7 and kc+16+khalf..+7,
    // matching A -> two 16B LDS reads per chunk. Two interleaved accumulator
    // chains hide WMMA accumulate latency.
    v8f acc0 = {};
    v8f acc1 = {};
#pragma unroll
    for (int c = 0; c < 16; c += 2) {
      const int kc0 = ks * 512 + c * 32;
      const int kc1 = kc0 + 32;
      v16bf b0 = __builtin_shufflevector(*(const v8bf*)(xbf + kc0 + khalf), *(const v8bf*)(xbf + kc0 + 16 + khalf), 0, 1, 2, 3, 4, 5, 6, 7, 8, 9, 10, 11, 12, 13, 14, 15);
      v16bf b1 = __builtin_shufflevector(*(const v8bf*)(xbf + kc1 + khalf), *(const v8bf*)(xbf + kc1 + 16 + khalf), 0, 1, 2, 3, 4, 5, 6, 7, 8, 9, 10, 11, 12, 13, 14, 15);
      acc0 = __builtin_amdgcn_wmma_f32_16x16x32_bf16(
                 false, a[c],     false, b0, (short)0, acc0, false, false);
      acc1 = __builtin_amdgcn_wmma_f32_16x16x32_bf16(
                 false, a[c + 1], false, b1, (short)0, acc1, false, false);
    }

    // C layout: VGPR r, lo-lanes M=r, hi-lanes M=8+r; all N columns equal.
    // Lane with N==0 in each half publishes its 8 y rows.
    if (m == 0) {
#pragma unroll
      for (int r = 0; r < 8; ++r)
        red[tile * 4 + ks][khalf + r] = acc0[r] + acc1[r];
    }
    __syncthreads();

    if (tid < ROWS_PER_WG) {
      const int tt = tid >> 4, mm = tid & 15;
      float y = red[tt * 4 + 0][mm] + red[tt * 4 + 1][mm]
              + red[tt * 4 + 2][mm] + red[tt * 4 + 3][mm];
      float xv = tanhf(wIn * u[t] + y);
      xnext[row0 + tid] = xv;
      dotbuf[tid] = xv * wOut;
    }
    __syncthreads();

    // Grid-wide step barrier (all 64 WGs resident) + deterministic partial dot
    if (tid == 0) {
      if (t >= WASHOUT) {
        float s = 0.0f;
#pragma unroll
        for (int i = 0; i < ROWS_PER_WG; ++i) s += dotbuf[i];
        partials[(size_t)(t - WASHOUT) * NBLK + wg] = s;
      }
      __threadfence();                       // release x writes
      atomicAdd(cnt, 1u);
      const unsigned int target = (unsigned int)(t + 1) * NBLK;
      while (__hip_atomic_load(cnt, __ATOMIC_ACQUIRE,
                               __HIP_MEMORY_SCOPE_AGENT) < target)
        __builtin_amdgcn_s_sleep(1);
    }
    __syncthreads();
    __threadfence();                         // acquire other WGs' x writes
  }
}

__global__ void esn_reduce(const float* __restrict__ partials,
                           float* __restrict__ out, int n) {
  int i = blockIdx.x * blockDim.x + threadIdx.x;
  if (i < n) {
    const float* p = partials + (size_t)i * NBLK;
    float s = 0.0f;
#pragma unroll
    for (int j = 0; j < NBLK; ++j) s += p[j];
    out[i] = s;
  }
}

extern "C" void kernel_launch(void* const* d_in, const int* in_sizes, int n_in,
                              void* d_out, int out_size, void* d_ws, size_t ws_size,
                              hipStream_t stream) {
  const float* u     = (const float*)d_in[0];
  const float* w_in  = (const float*)d_in[1];
  const float* w_res = (const float*)d_in[2];
  const float* w_out = (const float*)d_in[3];
  float* out = (float*)d_out;

  char* ws = (char*)d_ws;
  unsigned int* cnt = (unsigned int*)ws;
  float* xstate     = (float*)(ws + 256);
  float* partials   = (float*)(ws + 256 + 2 * HIDDEN * sizeof(float));

  esn_init<<<1, THREADS, 0, stream>>>(cnt, xstate);
  esn_recur<<<NBLK, THREADS, 0, stream>>>(u, w_in, w_res, w_out,
                                          xstate, partials, cnt);
  const int n = out_size;  // T - WASHOUT = 7992
  esn_reduce<<<(n + THREADS - 1) / THREADS, THREADS, 0, stream>>>(partials, out, n);
}
